// TransformerEncoder_13314398617957
// MI455X (gfx1250) — matrix operand order repaired and checked by reading, not compiled, not run on hardware
//
#include <hip/hip_runtime.h>

// ---------------------------------------------------------------------------
// TransformerEncoder for MI455X (gfx1250): all GEMMs via v_wmma_f32_16x16x32_f16
// Block tile 128x128, 8 waves, wave tile 32x64 (8 WMMA per K-step of 32).
// ---------------------------------------------------------------------------

typedef _Float16 half_t;
typedef __attribute__((ext_vector_type(16))) _Float16 v16h;
typedef __attribute__((ext_vector_type(8)))  _Float16 v8h;
typedef __attribute__((ext_vector_type(8)))  float    v8f;

#define B_ 8
#define S_ 1024
#define D_ 512
#define H_ 8
#define F_ 2048
#define EPS_ 1e-3f

__device__ __forceinline__ v16h mk16(v8h lo, v8h hi) {
    return __builtin_shufflevector(lo, hi, 0,1,2,3,4,5,6,7,8,9,10,11,12,13,14,15);
}

// ---- Generic batched GEMM: C = alpha * (A @ B) + bias ----------------------
// A: [batch via z/aDiv] MxK f16 row-major (lda)
// B: [batch via z%bMod] KxN f16 row-major (ldb)
// C: offset = (z/cDiv)*strideC1 + (z%cMod)*strideC2, leading dim ldc
// flags bit0: store f16 (else f32); bit1: store transposed (C[n*ldc + m])
// Requires M%128==0, N%128==0, K%32==0 (true for every GEMM in this model).
//
// CDNA5 WMMA 16-bit operand layouts (wave32):
//  A 16x32: lane<16 -> row M=lane, holds K[0..7] + K[16..23]; lane>=16 -> K[8..15]+K[24..31]
//  B 32x16: lane<16 -> col N=lane, holds K[0..7] + K[16..23]; lane>=16 -> N=lane-16, K[8..15]+K[24..31]
//  C 16x16: lane<16 -> N=lane, M=vgpr r; lane>=16 -> N=lane-16, M=8+r
__global__ __launch_bounds__(256) void gemm_wmma_f16_kernel(
    const half_t* __restrict__ A, long long strideA, int aDiv, int lda,
    const half_t* __restrict__ Bm, long long strideB, int bMod, int ldb,
    void* __restrict__ Cv, long long strideC1, int cDiv,
    long long strideC2, int cMod, int ldc,
    const float* __restrict__ bias, int biasMod, float alpha,
    int K, int Ncols, int flags)
{
    int z = blockIdx.z;
    const half_t* Ab = A + (long long)(z / aDiv) * strideA;
    const half_t* Bb = Bm + (long long)(z % bMod) * strideB;
    long long coff = (long long)(z / cDiv) * strideC1 + (long long)(z % cMod) * strideC2;

    int lane = threadIdx.x & 31;
    int wave = threadIdx.x >> 5;
    int wm = wave & 3;                 // 4 waves along M
    int wn = wave >> 2;                // 2 waves along N
    int mBase = blockIdx.x * 128 + wm * 32;
    int nBase = blockIdx.y * 128 + wn * 64;

    // Pointer-stepped fragment addresses (no per-iteration multiplies).
    const half_t* pA = Ab + (long long)(mBase + (lane & 15)) * lda + ((lane >> 4) << 3);
    const half_t* pB = Bb + (long long)((lane >> 4) << 3) * ldb + nBase + (lane & 15);
    const long long aRow16 = (long long)16 * lda;   // A fragment for rows mBase+16..31
    const long long bStep  = (long long)32 * ldb;   // K advance for B

    v8f zero = {0.f, 0.f, 0.f, 0.f, 0.f, 0.f, 0.f, 0.f};
    v8f acc[8];
#pragma unroll
    for (int t = 0; t < 8; ++t) acc[t] = zero;

    for (int k0 = 0; k0 < K; k0 += 32) {
        v16h a0 = mk16(*(const v8h*)(pA),          *(const v8h*)(pA + 16));
        v16h a1 = mk16(*(const v8h*)(pA + aRow16), *(const v8h*)(pA + aRow16 + 16));
        v16h bf[4];
#pragma unroll
        for (int j = 0; j < 4; ++j)
#pragma unroll
            for (int s = 0; s < 16; ++s)
                bf[j][s] = pB[(long long)((s & 7) + ((s >> 3) << 4)) * ldb + j * 16];

        // Speculative prefetch of next K-step (silently dropped past the end).
        __builtin_prefetch(pA + 32, 0, 1);
        __builtin_prefetch(pB + bStep, 0, 1);

#pragma unroll
        for (int j = 0; j < 4; ++j) {
            acc[j]     = __builtin_amdgcn_wmma_f32_16x16x32_f16(false, a0, false, bf[j], (short)0, acc[j],     false, false);
            acc[4 + j] = __builtin_amdgcn_wmma_f32_16x16x32_f16(false, a1, false, bf[j], (short)0, acc[4 + j], false, false);
        }
        pA += 32;
        pB += bStep;
    }

    int col   = lane & 15;
    int rbase = (lane >> 4) << 3;
    const float* bp = bias ? (bias + (long long)(z % biasMod) * Ncols) : (const float*)0;

#pragma unroll
    for (int t = 0; t < 8; ++t) {
        int i = t >> 2, j = t & 3;          // acc[t] = tile (i along M, j along N)
        int mg = mBase + i * 16 + rbase;
        int ng = nBase + j * 16 + col;
        float bv = bp ? bp[ng] : 0.f;
#pragma unroll
        for (int r = 0; r < 8; ++r) {
            float vv = acc[t][r] * alpha + bv;
            long long idx = (flags & 2) ? (coff + (long long)ng * ldc + (mg + r))
                                        : (coff + (long long)(mg + r) * ldc + ng);
            if (flags & 1) ((half_t*)Cv)[idx] = (half_t)vv;
            else           ((float*)Cv)[idx]  = vv;
        }
    }
}

// ---- f32 -> f16 convert ----------------------------------------------------
__global__ __launch_bounds__(256) void f32_to_f16_kernel(const float* __restrict__ x,
                                                         half_t* __restrict__ y, long long n) {
    long long i = (long long)blockIdx.x * 256 + threadIdx.x;
    if (i < n) y[i] = (half_t)x[i];
}

// ---- row softmax over S_ columns; one wave per row -------------------------
__global__ __launch_bounds__(256) void softmax_rows_kernel(const float* __restrict__ Sc,
                                                           half_t* __restrict__ attn) {
    int lane = threadIdx.x & 31;
    int wave = threadIdx.x >> 5;
    long long row = (long long)blockIdx.x * 8 + wave;
    const float* p = Sc + row * (long long)S_;
    float v[32];
    float mx = -3.4e38f;
#pragma unroll
    for (int i = 0; i < 32; ++i) { v[i] = p[i * 32 + lane]; mx = fmaxf(mx, v[i]); }
#pragma unroll
    for (int d = 16; d >= 1; d >>= 1) mx = fmaxf(mx, __shfl_xor(mx, d, 32));
    float sum = 0.f;
#pragma unroll
    for (int i = 0; i < 32; ++i) { v[i] = __expf(v[i] - mx); sum += v[i]; }
#pragma unroll
    for (int d = 16; d >= 1; d >>= 1) sum += __shfl_xor(sum, d, 32);
    float inv = 1.f / sum;
    half_t* q = attn + row * (long long)S_;
#pragma unroll
    for (int i = 0; i < 32; ++i) q[i * 32 + lane] = (half_t)(v[i] * inv);
}

// ---- residual add + LayerNorm over D_ columns; one wave per row ------------
__global__ __launch_bounds__(256) void add_ln_kernel(const float* __restrict__ X,
                                                     const float* __restrict__ R,
                                                     const float* __restrict__ g,
                                                     const float* __restrict__ b,
                                                     float* __restrict__ out,
                                                     half_t* __restrict__ out16) {
    int lane = threadIdx.x & 31;
    int wave = threadIdx.x >> 5;
    long long row = (long long)blockIdx.x * 8 + wave;
    const float* x = X + row * (long long)D_;
    const float* r = R + row * (long long)D_;
    float v[16];
    float s = 0.f;
#pragma unroll
    for (int i = 0; i < 16; ++i) { int idx = i * 32 + lane; v[i] = x[idx] + r[idx]; s += v[i]; }
#pragma unroll
    for (int d = 16; d >= 1; d >>= 1) s += __shfl_xor(s, d, 32);
    float mean = s * (1.f / D_);
    float s2 = 0.f;
#pragma unroll
    for (int i = 0; i < 16; ++i) { float dd = v[i] - mean; s2 += dd * dd; }
#pragma unroll
    for (int d = 16; d >= 1; d >>= 1) s2 += __shfl_xor(s2, d, 32);
    float inv = rsqrtf(s2 * (1.f / D_) + EPS_);
#pragma unroll
    for (int i = 0; i < 16; ++i) {
        int idx = i * 32 + lane;
        float y = (v[i] - mean) * inv * g[idx] + b[idx];
        out[row * (long long)D_ + idx] = y;
        if (out16) out16[row * (long long)D_ + idx] = (half_t)y;
    }
}

// ---- BatchNorm (moving stats) + PReLU, write f16 ---------------------------
__global__ __launch_bounds__(256) void bn_prelu_kernel(const float* __restrict__ Hv,
                                                       const float* __restrict__ mm,
                                                       const float* __restrict__ vv,
                                                       const float* __restrict__ gg,
                                                       const float* __restrict__ bb,
                                                       const float* __restrict__ aa,
                                                       half_t* __restrict__ out) {
    long long i = (long long)blockIdx.x * 256 + threadIdx.x;
    int f = (int)(i & (long long)(F_ - 1));
    long long row = i >> 11;               // F_ == 2048 == 2^11
    int s = (int)(row & (long long)(S_ - 1));
    float y = (Hv[i] - mm[f]) * rsqrtf(vv[f] + EPS_) * gg[f] + bb[f];
    float al = aa[(long long)s * F_ + f];
    out[i] = (half_t)(y > 0.f ? y : al * y);
}

// ---------------------------------------------------------------------------
extern "C" void kernel_launch(void* const* d_in, const int* in_sizes, int n_in,
                              void* d_out, int out_size, void* d_ws, size_t ws_size,
                              hipStream_t stream) {
    (void)in_sizes; (void)n_in; (void)out_size; (void)ws_size;

    const float* Qf  = (const float*)d_in[0];
    const float* Kf  = (const float*)d_in[1];
    const float* Vf  = (const float*)d_in[2];
    const float* WQ  = (const float*)d_in[3];
    const float* bQ  = (const float*)d_in[4];
    const float* WK  = (const float*)d_in[5];
    const float* bK  = (const float*)d_in[6];
    const float* WV  = (const float*)d_in[7];
    const float* bV  = (const float*)d_in[8];
    const float* Wo  = (const float*)d_in[9];
    const float* bo  = (const float*)d_in[10];
    const float* W0  = (const float*)d_in[11];
    const float* b0  = (const float*)d_in[12];
    const float* g0  = (const float*)d_in[13];
    const float* be0 = (const float*)d_in[14];
    const float* m0  = (const float*)d_in[15];
    const float* v0  = (const float*)d_in[16];
    const float* a0  = (const float*)d_in[17];
    const float* W1  = (const float*)d_in[18];
    const float* b1  = (const float*)d_in[19];
    const float* g1  = (const float*)d_in[20];
    const float* be1 = (const float*)d_in[21];
    const float* m1  = (const float*)d_in[22];
    const float* v1  = (const float*)d_in[23];
    const float* a1  = (const float*)d_in[24];
    const float* W2  = (const float*)d_in[25];
    const float* b2  = (const float*)d_in[26];
    const float* ln0g = (const float*)d_in[27];
    const float* ln0b = (const float*)d_in[28];
    const float* ln1g = (const float*)d_in[29];
    const float* ln1b = (const float*)d_in[30];
    float* outp = (float*)d_out;

    // ---- workspace layout (bump allocator with static reuse) ----
    char* ws = (char*)d_ws;
    size_t o = 0;
    auto take = [&](size_t bytes) { size_t r = o; o += (bytes + 255) & ~(size_t)255; return r; };

    const long long nQKV  = (long long)B_ * S_ * D_;        //  4 Mi elems
    const long long nWqkv = (long long)H_ * D_ * D_;        //  2 Mi
    const long long nW0e  = (long long)D_ * F_;             //  1 Mi
    const long long nW1e  = (long long)F_ * F_;             //  4 Mi
    const long long nW2e  = (long long)F_ * D_;             //  1 Mi
    const long long nQH   = (long long)B_ * H_ * S_ * D_;   // 32 Mi
    const long long nSc   = (long long)B_ * H_ * S_ * S_;   // 64 Mi
    const long long nHid  = (long long)B_ * S_ * F_;        // 16 Mi

    half_t* WQb = (half_t*)(ws + take(nWqkv * 2));
    half_t* WKb = (half_t*)(ws + take(nWqkv * 2));
    half_t* WVb = (half_t*)(ws + take(nWqkv * 2));
    half_t* Wob = (half_t*)(ws + take(nWqkv * 2));          // [H*D, D] = 2 Mi
    half_t* W0b = (half_t*)(ws + take(nW0e * 2));
    half_t* W1b = (half_t*)(ws + take(nW1e * 2));
    half_t* W2b = (half_t*)(ws + take(nW2e * 2));
    half_t* Qb  = (half_t*)(ws + take(nQKV * 2));
    half_t* Kb  = (half_t*)(ws + take(nQKV * 2));
    half_t* Vb  = (half_t*)(ws + take(nQKV * 2));
    half_t* qh  = (half_t*)(ws + take(nQH * 2));            // 64 MiB
    half_t* khT = (half_t*)(ws + take(nQH * 2));            // 64 MiB (contiguous after qh)
    half_t* vh  = (half_t*)(ws + take(nQH * 2));            // 64 MiB
    char* uni   = ws + take((size_t)nSc * 4);               // 256 MiB union region
    float*  mha  = (float*)(ws + take(nQKV * 4));
    float*  xbuf = (float*)(ws + take(nQKV * 4));
    half_t* xb   = (half_t*)(ws + take(nQKV * 2));
    float*  ffb  = (float*)(ws + take(nQKV * 4));

    // reuse: attn(128 MiB) overlays dead qh+khT; cat/h-f32 inside dead scores
    float*  scores = (float*)uni;                            // [B,H,S,S] f32
    half_t* cat    = (half_t*)uni;                           // [B,S,H*D] f16 (scores dead)
    float*  hbuf   = (float*)(uni + (size_t)134217728);      // [B*S,F] f32 @ +128 MiB
    half_t* attn   = (half_t*)qh;                            // [B,H,S,S] f16
    half_t* h0b    = (half_t*)vh;                            // [B*S,F] f16 (vh dead)
    half_t* h1b    = (half_t*)((char*)vh + (size_t)nHid * 2);

    // ---- convert inputs/weights to f16 ----
    auto cvt = [&](const float* s, half_t* d, long long n) {
        f32_to_f16_kernel<<<dim3((unsigned)((n + 255) / 256)), dim3(256), 0, stream>>>(s, d, n);
    };
    cvt(WQ, WQb, nWqkv); cvt(WK, WKb, nWqkv); cvt(WV, WVb, nWqkv);
    cvt(Wo, Wob, nWqkv); cvt(W0, W0b, nW0e); cvt(W1, W1b, nW1e); cvt(W2, W2b, nW2e);
    cvt(Qf, Qb, nQKV);   cvt(Kf, Kb, nQKV); cvt(Vf, Vb, nQKV);

    auto gemm = [&](const half_t* A, long long sA, int aDiv, int lda,
                    const half_t* Bmm, long long sB, int bMod, int ldb,
                    void* C, long long sC1, int cDiv, long long sC2, int cMod, int ldc,
                    const float* bias, int biasMod, float alpha,
                    int M, int N, int K, int batch, int flags) {
        dim3 grid(M / 128, N / 128, batch);
        gemm_wmma_f16_kernel<<<grid, dim3(256), 0, stream>>>(
            A, sA, aDiv, lda, Bmm, sB, bMod, ldb,
            C, sC1, cDiv, sC2, cMod, ldc, bias, biasMod, alpha, K, N, flags);
    };

    const long long SD = (long long)S_ * D_;   // 524288
    const long long DD = (long long)D_ * D_;   // 262144
    const long long SS = (long long)S_ * S_;   // 1048576

    // ---- per-head projections: qh[b,h]=Q[b]@WQ[h]+bQ[h], etc. ----
    gemm(Qb, SD, H_, D_, WQb, DD, H_, D_, qh,  SD, 1, 0, 1, D_, bQ, H_, 1.f, S_, D_, D_, B_ * H_, 1);
    // kh stored TRANSPOSED -> khT[b,h] is [D,S]  (flags=1|2)
    gemm(Kb, SD, H_, D_, WKb, DD, H_, D_, khT, SD, 1, 0, 1, S_, bK, H_, 1.f, S_, D_, D_, B_ * H_, 3);
    gemm(Vb, SD, H_, D_, WVb, DD, H_, D_, vh,  SD, 1, 0, 1, D_, bV, H_, 1.f, S_, D_, D_, B_ * H_, 1);

    // ---- scores = (qh @ khT) * (1/D), f32 ----
    gemm(qh, SD, 1, D_, khT, SD, B_ * H_, S_, scores, SS, 1, 0, 1, S_,
         (const float*)0, 1, 1.0f / (float)D_, S_, S_, D_, B_ * H_, 0);

    // ---- softmax rows -> attn f16 (overlays qh/khT, now dead) ----
    softmax_rows_kernel<<<dim3((unsigned)((long long)B_ * H_ * S_ / 8)), dim3(256), 0, stream>>>(scores, attn);

    // ---- ctx = attn @ vh, written directly into cat layout [B,S,H*D] ----
    gemm(attn, SS, 1, S_, vh, SD, B_ * H_, D_, cat,
         (long long)S_ * H_ * D_, H_, (long long)D_, H_, H_ * D_,
         (const float*)0, 1, 1.f, S_, D_, S_, B_ * H_, 1);

    // ---- mha = cat @ Wo + bo ----
    gemm(cat, 0, 1, H_ * D_, Wob, 0, 1, D_, mha, 0, 1, 0, 1, D_,
         bo, 1, 1.f, B_ * S_, D_, H_ * D_, 1, 0);

    // ---- x = LayerNorm(Q + mha) ; also f16 copy for next GEMM ----
    add_ln_kernel<<<dim3(B_ * S_ / 8), dim3(256), 0, stream>>>(Qf, mha, ln0g, ln0b, xbuf, xb);

    // ---- FeedForward ----
    gemm(xb, 0, 1, D_, W0b, 0, 1, F_, hbuf, 0, 1, 0, 1, F_, b0, 1, 1.f, B_ * S_, F_, D_, 1, 0);
    bn_prelu_kernel<<<dim3((unsigned)(nHid / 256)), dim3(256), 0, stream>>>(hbuf, m0, v0, g0, be0, a0, h0b);

    gemm(h0b, 0, 1, F_, W1b, 0, 1, F_, hbuf, 0, 1, 0, 1, F_, b1, 1, 1.f, B_ * S_, F_, F_, 1, 0);
    bn_prelu_kernel<<<dim3((unsigned)(nHid / 256)), dim3(256), 0, stream>>>(hbuf, m1, v1, g1, be1, a1, h1b);

    gemm(h1b, 0, 1, F_, W2b, 0, 1, D_, ffb, 0, 1, 0, 1, D_, b2, 1, 1.f, B_ * S_, D_, F_, 1, 0);

    // ---- out = LayerNorm(x + ff) ----
    add_ln_kernel<<<dim3(B_ * S_ / 8), dim3(256), 0, stream>>>(xbuf, ffb, ln1g, ln1b, outp, (half_t*)0);
}
